// DARTSVQBlock_21174188769556
// MI455X (gfx1250) — matrix operand order repaired and checked by reading, not compiled, not run on hardware
//
#include <hip/hip_runtime.h>
#include <hip/hip_bf16.h>
#include <math.h>

typedef __attribute__((ext_vector_type(16))) _Float16 v16h;
typedef __attribute__((ext_vector_type(8)))  float    v8f;

#define D_EMBED  256
#define N_ROWS   65536
#define N_TILES  60      /* 960 cols / 16 */
#define KCHUNKS  8       /* 256 / 32     */

/* workspace byte offsets (all 64B+ aligned) */
#define WS_ALPHA_OFF   0u        /* 4 floats  */
#define WS_SSE_OFF     64u       /* 4 floats  */
#define WS_CN_OFF      256u      /* 960 floats: per-column squared norms          */
#define WS_DICT_OFF    4096u     /* 960*256 floats: transposed dicts (col-major)  */
#define WS_FRAGB_OFF   987136u   /* 60*8*32*16 halves: WMMA B fragments           */

/* ---------- prep: softmax(alpha_vq) + zero sse accumulators ---------- */
__global__ void vq_prep_alpha(const float* __restrict__ av,
                              float* __restrict__ wsAlpha,
                              float* __restrict__ wsSse) {
  if (threadIdx.x == 0 && blockIdx.x == 0) {
    float a0 = av[0], a1 = av[1], a2 = av[2], a3 = av[3];
    float m = fmaxf(fmaxf(a0, a1), fmaxf(a2, a3));
    float e0 = expf(a0 - m), e1 = expf(a1 - m), e2 = expf(a2 - m), e3 = expf(a3 - m);
    float inv = 1.0f / (e0 + e1 + e2 + e3);
    wsAlpha[0] = e0 * inv; wsAlpha[1] = e1 * inv;
    wsAlpha[2] = e2 * inv; wsAlpha[3] = e3 * inv;
    wsSse[0] = 0.f; wsSse[1] = 0.f; wsSse[2] = 0.f; wsSse[3] = 0.f;
  }
}

/* ---------- prep: column norms + transposed f32 dictionaries ---------- */
__global__ void vq_prep_dict(const float* __restrict__ d6, const float* __restrict__ d7,
                             const float* __restrict__ d8, const float* __restrict__ d9,
                             float* __restrict__ dicT, float* __restrict__ colNorm) {
  int c = blockIdx.x * blockDim.x + threadIdx.x;
  if (c >= 960) return;
  int dId  = (c < 64) ? 0 : (c < 192) ? 1 : (c < 448) ? 2 : 3;
  int base = (dId == 0) ? 0 : (dId == 1) ? 64 : (dId == 2) ? 192 : 448;
  int K    = 64 << dId;
  const float* dp = (dId == 0) ? d6 : (dId == 1) ? d7 : (dId == 2) ? d8 : d9;
  int local = c - base;
  float nrm = 0.f;
  float* dst = dicT + (size_t)c * D_EMBED;
  for (int d = 0; d < D_EMBED; ++d) {
    float v = dp[(size_t)d * K + local];
    dst[d] = v;
    nrm = fmaf(v, v, nrm);
  }
  colNorm[c] = nrm;
}

/* ---------- prep: pack f16 B fragments in WMMA lane/VGPR order ---------- */
__global__ void vq_prep_fragB(const float* __restrict__ d6, const float* __restrict__ d7,
                              const float* __restrict__ d8, const float* __restrict__ d9,
                              _Float16* __restrict__ fragB) {
  int g = blockIdx.x * blockDim.x + threadIdx.x;
  if (g >= N_TILES * KCHUNKS * 32 * 16) return;
  int t    = g >> 12;
  int rem  = g & 4095;
  int c    = rem >> 9;
  int rem2 = rem & 511;
  int lane = rem2 >> 4;
  int h    = rem2 & 15;
  int colT = lane & 15;
  int k    = c * 32 + h + ((lane >> 4) << 4);
  int dId  = (t < 4) ? 0 : (t < 12) ? 1 : (t < 28) ? 2 : 3;
  int tb   = (dId == 0) ? 0 : (dId == 1) ? 4 : (dId == 2) ? 12 : 28;
  int K    = 64 << dId;
  const float* dp = (dId == 0) ? d6 : (dId == 1) ? d7 : (dId == 2) ? d8 : d9;
  int colInDict = (t - tb) * 16 + colT;
  fragB[g] = (_Float16)dp[(size_t)k * K + colInDict];
}

/* ---------- CDNA5 async memory->LDS helpers ---------- */
__device__ __forceinline__ void async_copy_32B(unsigned ldsDst, const _Float16* g) {
  /* two ASYNCcnt-tracked b128 copies, no VGPR data path */
  asm volatile("global_load_async_to_lds_b128 %0, %1, off\n\t"
               "global_load_async_to_lds_b128 %2, %3, off"
               :
               : "v"(ldsDst), "v"(g), "v"(ldsDst + 16u), "v"(g + 8)
               : "memory");
}
__device__ __forceinline__ void wait_async0() {
  asm volatile("s_wait_asynccnt 0" ::: "memory");
}

/* ---------- main: WMMA distances + argmin + gather + weighted output ---------- */
__global__ __launch_bounds__(256)
void vq_main(const float* __restrict__ x, const _Float16* __restrict__ fragB,
             const float* __restrict__ colNorm, const float* __restrict__ dicT,
             const float* __restrict__ wsAlpha, float* __restrict__ wsSse,
             float* __restrict__ out) {
  __shared__ __attribute__((aligned(32))) _Float16 ldsB[2][KCHUNKS * 32 * 16]; /* 2x8KB */
  __shared__ int ldsIdx[8][4][16];

  const int tid  = threadIdx.x;
  const int w    = tid >> 5;
  const int lane = tid & 31;
  const int m0   = lane & 15;
  const int hi   = lane >> 4;
  const int rowBase = blockIdx.x * 128 + w * 16;

  /* LDS byte offsets of the two staging buffers (low 32 bits of flat address) */
  const unsigned buf0 = (unsigned)(uintptr_t)(&ldsB[0][0]) + (unsigned)tid * 32u;
  const unsigned buf1 = (unsigned)(uintptr_t)(&ldsB[1][0]) + (unsigned)tid * 32u;

  /* ---- load A fragments: 16 rows x 256 K, f32->f16, contiguous k order (16*hi + s) ---- */
  const float* xrow = x + (size_t)(rowBase + m0) * D_EMBED;
  v16h a[KCHUNKS];
#pragma unroll
  for (int c = 0; c < KCHUNKS; ++c) {
#pragma unroll
    for (int v = 0; v < 8; ++v) {
      int k = c * 32 + 2 * v + (hi ? 16 : 0);
      float2 f = *(const float2*)(xrow + k);
      a[c][2 * v]     = (_Float16)f.x;
      a[c][2 * v + 1] = (_Float16)f.y;
    }
  }

  float best[8]; int bidx[8];
#pragma unroll
  for (int r = 0; r < 8; ++r) { best[r] = 3.0e38f; bidx[r] = 0; }

  /* prologue: async-stage tile 0 into buffer 0 */
  async_copy_32B(buf0, fragB + (size_t)tid * 16);
  wait_async0();
  __syncthreads();

  int cur = 0;   /* which buffer holds the current tile */
  int kd  = 0;   /* current dictionary */
  int tb  = 0;   /* first tile of current dictionary */

  for (int t = 0; t < N_TILES; ++t) {
    /* prefetch next tile into the other buffer while we compute */
    if (t + 1 < N_TILES) {
      unsigned dst = cur ? buf0 : buf1;
      async_copy_32B(dst, fragB + (size_t)(t + 1) * 4096 + tid * 16);
    }

    const v16h* bv = (const v16h*)(cur ? &ldsB[1][0] : &ldsB[0][0]);
    v8f cs0 = {}, cs1 = {};
#pragma unroll
    for (int c = 0; c < KCHUNKS; c += 2) {
      v16h b0 = bv[c * 32 + lane];
      v16h b1 = bv[(c + 1) * 32 + lane];
      cs0 = __builtin_amdgcn_wmma_f32_16x16x32_f16(
          false, a[c], false, b0, (short)0, cs0, false, false);
      cs1 = __builtin_amdgcn_wmma_f32_16x16x32_f16(
          false, a[c + 1], false, b1, (short)0, cs1, false, false);
    }

    const float cn = colNorm[t * 16 + m0];
    const int localcol = (t - tb) * 16 + m0;
#pragma unroll
    for (int r = 0; r < 8; ++r) {
      float d = fmaf(-2.0f, cs0[r] + cs1[r], cn);
      if (d < best[r]) { best[r] = d; bidx[r] = localcol; }
    }

    /* dictionary boundary: reduce argmin over columns, emit indices, reset */
    if (t == 3 || t == 11 || t == 27 || t == 59) {
#pragma unroll
      for (int r = 0; r < 8; ++r) {
#pragma unroll
        for (int off = 8; off >= 1; off >>= 1) {
          float ov = __shfl_xor(best[r], off, 16);
          int   oi = __shfl_xor(bidx[r], off, 16);
          if (ov < best[r] || (ov == best[r] && oi < bidx[r])) {
            best[r] = ov; bidx[r] = oi;
          }
        }
        if (m0 == r) ldsIdx[w][kd][r + 8 * hi] = bidx[r];
        best[r] = 3.0e38f; bidx[r] = 0;
      }
      kd++; tb = t + 1;
    }

    if (t + 1 < N_TILES) {
      wait_async0();       /* next tile landed in LDS (this wave's copies) */
      __syncthreads();     /* everyone done with current tile + all copies visible */
      cur ^= 1;
    }
  }
  __syncthreads();

  /* ---- gather + weighted output + SSE (coalesced: all lanes on one row) ---- */
  const float al0 = wsAlpha[0], al1 = wsAlpha[1], al2 = wsAlpha[2], al3 = wsAlpha[3];
  float s0 = 0.f, s1 = 0.f, s2 = 0.f, s3 = 0.f;

  for (int m = 0; m < 16; ++m) {
    const int i0 = ldsIdx[w][0][m], i1 = ldsIdx[w][1][m];
    const int i2 = ldsIdx[w][2][m], i3 = ldsIdx[w][3][m];
    const float* q0 = dicT + (size_t)(0   + i0) * D_EMBED + lane * 8;
    const float* q1 = dicT + (size_t)(64  + i1) * D_EMBED + lane * 8;
    const float* q2 = dicT + (size_t)(192 + i2) * D_EMBED + lane * 8;
    const float* q3 = dicT + (size_t)(448 + i3) * D_EMBED + lane * 8;
    const float* xr = x   + (size_t)(rowBase + m) * D_EMBED + lane * 8;
    float*       orow = out + (size_t)(rowBase + m) * D_EMBED + lane * 8;

#pragma unroll
    for (int part = 0; part < 2; ++part) {
      const int p = part * 4;
      float4 xv = *(const float4*)(xr + p);
      float4 v0 = *(const float4*)(q0 + p);
      float4 v1 = *(const float4*)(q1 + p);
      float4 v2 = *(const float4*)(q2 + p);
      float4 v3 = *(const float4*)(q3 + p);
      float4 wq;
      wq.x = al0 * v0.x + al1 * v1.x + al2 * v2.x + al3 * v3.x;
      wq.y = al0 * v0.y + al1 * v1.y + al2 * v2.y + al3 * v3.y;
      wq.z = al0 * v0.z + al1 * v1.z + al2 * v2.z + al3 * v3.z;
      wq.w = al0 * v0.w + al1 * v1.w + al2 * v2.w + al3 * v3.w;
      *(float4*)(orow + p) = wq;

      float dx;
      dx = xv.x - v0.x; s0 = fmaf(dx, dx, s0);
      dx = xv.y - v0.y; s0 = fmaf(dx, dx, s0);
      dx = xv.z - v0.z; s0 = fmaf(dx, dx, s0);
      dx = xv.w - v0.w; s0 = fmaf(dx, dx, s0);
      dx = xv.x - v1.x; s1 = fmaf(dx, dx, s1);
      dx = xv.y - v1.y; s1 = fmaf(dx, dx, s1);
      dx = xv.z - v1.z; s1 = fmaf(dx, dx, s1);
      dx = xv.w - v1.w; s1 = fmaf(dx, dx, s1);
      dx = xv.x - v2.x; s2 = fmaf(dx, dx, s2);
      dx = xv.y - v2.y; s2 = fmaf(dx, dx, s2);
      dx = xv.z - v2.z; s2 = fmaf(dx, dx, s2);
      dx = xv.w - v2.w; s2 = fmaf(dx, dx, s2);
      dx = xv.x - v3.x; s3 = fmaf(dx, dx, s3);
      dx = xv.y - v3.y; s3 = fmaf(dx, dx, s3);
      dx = xv.z - v3.z; s3 = fmaf(dx, dx, s3);
      dx = xv.w - v3.w; s3 = fmaf(dx, dx, s3);
    }
  }

#pragma unroll
  for (int off = 16; off >= 1; off >>= 1) {
    s0 += __shfl_xor(s0, off, 32);
    s1 += __shfl_xor(s1, off, 32);
    s2 += __shfl_xor(s2, off, 32);
    s3 += __shfl_xor(s3, off, 32);
  }
  if (lane == 0) {
    atomicAdd(&wsSse[0], s0);
    atomicAdd(&wsSse[1], s1);
    atomicAdd(&wsSse[2], s2);
    atomicAdd(&wsSse[3], s3);
  }
}

/* ---------- finalize scalar loss ---------- */
__global__ void vq_finalize(const float* __restrict__ wsAlpha,
                            const float* __restrict__ wsSse,
                            float* __restrict__ out) {
  if (threadIdx.x == 0 && blockIdx.x == 0) {
    float s = wsAlpha[0] * wsSse[0] + wsAlpha[1] * wsSse[1] +
              wsAlpha[2] * wsSse[2] + wsAlpha[3] * wsSse[3];
    out[(size_t)N_ROWS * D_EMBED] = 1.25f * s * (1.0f / ((float)N_ROWS * (float)D_EMBED));
  }
}

extern "C" void kernel_launch(void* const* d_in, const int* in_sizes, int n_in,
                              void* d_out, int out_size, void* d_ws, size_t ws_size,
                              hipStream_t stream) {
  const float* x  = (const float*)d_in[0];
  const float* av = (const float*)d_in[1];
  const float* d6 = (const float*)d_in[2];
  const float* d7 = (const float*)d_in[3];
  const float* d8 = (const float*)d_in[4];
  const float* d9 = (const float*)d_in[5];

  char* ws = (char*)d_ws;
  float*    wsAlpha = (float*)(ws + WS_ALPHA_OFF);
  float*    wsSse   = (float*)(ws + WS_SSE_OFF);
  float*    wsCN    = (float*)(ws + WS_CN_OFF);
  float*    wsDicT  = (float*)(ws + WS_DICT_OFF);
  _Float16* wsFragB = (_Float16*)(ws + WS_FRAGB_OFF);
  float*    out     = (float*)d_out;

  vq_prep_alpha<<<1, 32, 0, stream>>>(av, wsAlpha, wsSse);
  vq_prep_dict<<<4, 256, 0, stream>>>(d6, d7, d8, d9, wsDicT, wsCN);
  vq_prep_fragB<<<960, 256, 0, stream>>>(d6, d7, d8, d9, wsFragB);
  vq_main<<<N_ROWS / 128, 256, 0, stream>>>(x, wsFragB, wsCN, wsDicT,
                                            wsAlpha, wsSse, out);
  vq_finalize<<<1, 32, 0, stream>>>(wsAlpha, wsSse, out);
}
